// Fourier_Head_57432302682195
// MI455X (gfx1250) — matrix operand order repaired and checked by reading, not compiled, not run on hardware
//
#include <hip/hip_runtime.h>
#include <hip/hip_bf16.h>
#include <math.h>

// ---------- types ----------
typedef __attribute__((ext_vector_type(8)))  float  v8f;
typedef __attribute__((ext_vector_type(16))) __bf16 v16bf;
typedef __attribute__((ext_vector_type(8)))  __bf16 v8bf;

#define NUM_FREQ   64
#define LQ         65          // L = NUM_FREQ + 1
#define TWO_L      130
#define NPAD       144         // 130 padded to 9 tiles of 16
#define DIN        1024
#define DOUT       512
#define K2         128         // [c_re(64) | -c_im(64)]
#define BROWS      65536       // 16*4096

union BFrag { v16bf v; v8bf h[2]; };

// ---------- prep: W (f32) -> bf16, zero-padded to 144 rows ----------
__global__ __launch_bounds__(256) void k_prep_w(const float* __restrict__ W,
                                                __bf16* __restrict__ Wbf) {
    int idx = blockIdx.x * 256 + threadIdx.x;       // 144*1024 elements
    int n = idx >> 10, k = idx & 1023;
    float v = (n < TWO_L) ? W[n * DIN + k] : 0.0f;
    Wbf[idx] = (__bf16)v;
}

// ---------- prep: trig table Tt[d, f] : f<64 -> cos, f>=64 -> sin ----------
__global__ __launch_bounds__(256) void k_prep_trig(__bf16* __restrict__ Tt) {
    int idx = blockIdx.x * 256 + threadIdx.x;       // 512*128 elements
    int d = idx >> 7, f = idx & 127;
    float center = -1.0f + (2.0f * (float)d + 1.0f) * (1.0f / (float)DOUT);
    float freq   = 3.14159265358979323846f * (float)((f & 63) + 1);
    float ph = center * freq;
    Tt[idx] = (__bf16)((f < 64) ? cosf(ph) : sinf(ph));
}

// ---------- fused: p = x@W^T + b (bf16 WMMA) -> LDS -> autocorr -> cmat ----------
// grid: 512 blocks x 256 threads; each wave owns one 16-row tile.
__global__ __launch_bounds__(256) void k_gemm_autocorr(
    const float*  __restrict__ x,    // [B, 1024] f32
    const __bf16* __restrict__ Wbf,  // [144, 1024] bf16
    const float*  __restrict__ bias, // [130] f32
    __bf16*       __restrict__ cmat) // [B, 128] bf16 = [c_re | -c_im]
{
    __shared__ float smp[8][16][134];   // interleaved complex p per wave
    __shared__ float sminv[8][16];      // 1 / r0 per row

    const int lane = threadIdx.x & 31;
    const int w    = threadIdx.x >> 5;
    const int hi   = lane >> 4;         // 0 | 1
    const int l15  = lane & 15;
    const long row0 = ((long)blockIdx.x * 8 + w) * 16;

    v8f acc[9];
    #pragma unroll
    for (int t = 0; t < 9; ++t) acc[t] = (v8f){0,0,0,0,0,0,0,0};

    // A fragment source: M = l15, K-chunks at +0/+16 (+8 for upper half-wave)
    const float* xrow = x + (row0 + l15) * DIN + hi * 8;

    for (int k0 = 0; k0 < DIN; k0 += 32) {
        __builtin_prefetch(xrow + k0 + 64, 0, 1);   // global_prefetch_b8
        v8f x0 = *(const v8f*)(xrow + k0);
        v8f x1 = *(const v8f*)(xrow + k0 + 16);
        BFrag a;
        #pragma unroll
        for (int e = 0; e < 8; ++e) {
            a.v[e]     = (__bf16)x0[e];
            a.v[e + 8] = (__bf16)x1[e];
        }
        #pragma unroll
        for (int t = 0; t < 9; ++t) {
            // B fragment: N = t*16 + l15, K-chunks at +0/+16 (+8 for upper half-wave)
            const __bf16* wp = Wbf + (t * 16 + l15) * DIN + k0 + hi * 8;
            BFrag bfr;
            bfr.h[0] = *(const v8bf*)(wp);
            bfr.h[1] = *(const v8bf*)(wp + 16);
            acc[t] = __builtin_amdgcn_wmma_f32_16x16x32_bf16(
                false, a.v, false, bfr.v, (short)0, acc[t], false, false);
        }
    }

    // bias + spill p tile to LDS as interleaved complex (re at 2j, im at 2j+1)
    #pragma unroll
    for (int t = 0; t < 9; ++t) {
        int col = t * 16 + l15;
        if (col < TWO_L) {
            float bb  = bias[col];
            int   pos = (col < LQ) ? (2 * col) : (2 * (col - LQ) + 1);
            #pragma unroll
            for (int v = 0; v < 8; ++v)
                smp[w][v + hi * 8][pos] = acc[t][v] + bb;   // C layout: M=v(+8), N=l15
        }
    }

    // r0 = sum |a|^2 per row (lanes 0..15, one row each)
    if (lane < 16) {
        float r0 = 0.0f;
        #pragma unroll 1
        for (int j = 0; j < LQ; ++j) {
            float2 aj = *(const float2*)(&smp[w][lane][2 * j]);
            r0 += aj.x * aj.x + aj.y * aj.y;
        }
        sminv[w][lane] = 1.0f / r0;
    }

    // autocorrelation: 16 rows x 64 lags distributed over 32 lanes
    #pragma unroll 1
    for (int task = lane; task < 16 * 64; task += 32) {
        int row = task >> 6;
        int k   = (task & 63) + 1;
        float rre = 0.0f, rim = 0.0f;
        const float* pr = &smp[w][row][0];
        #pragma unroll 1
        for (int j = 0; j + k < LQ; ++j) {
            float2 aj  = *(const float2*)(pr + 2 * j);
            float2 ajk = *(const float2*)(pr + 2 * (j + k));
            rre += ajk.x * aj.x + ajk.y * aj.y;   // Re(a[j+k] * conj(a[j]))
            rim += ajk.y * aj.x - ajk.x * aj.y;   // Im(a[j+k] * conj(a[j]))
        }
        float inv = sminv[w][row];
        long  o   = (row0 + row) * K2;
        cmat[o + (k - 1)]      = (__bf16)( rre * inv);   // c.re
        cmat[o + 64 + (k - 1)] = (__bf16)(-rim * inv);   // -c.im
    }
}

// ---------- like = 0.5 + cmat @ Tt^T ; normalize ; log ----------
// grid: 4096 blocks x 256 threads; 16 rows x 512 cols per block.
__global__ __launch_bounds__(256) void k_trig(
    const __bf16* __restrict__ cmat,  // [B, 128] bf16
    const __bf16* __restrict__ Tt,    // [512, 128] bf16
    float*        __restrict__ out)   // [B, 512] f32
{
    __shared__ float smlike[16][516];
    __shared__ float smpart[16][16];
    __shared__ float sminv[16];

    const int tid  = threadIdx.x;
    const int lane = tid & 31;
    const int w    = tid >> 5;
    const int hi   = lane >> 4;
    const int l15  = lane & 15;
    const long row0 = (long)blockIdx.x * 16;

    v8f acc[4];
    #pragma unroll
    for (int i = 0; i < 4; ++i) acc[i] = (v8f){0,0,0,0,0,0,0,0};

    const __bf16* arow = cmat + (row0 + l15) * K2 + hi * 8;
    #pragma unroll
    for (int ks = 0; ks < 4; ++ks) {
        int k0 = ks * 32;
        BFrag a;
        a.h[0] = *(const v8bf*)(arow + k0);
        a.h[1] = *(const v8bf*)(arow + k0 + 16);
        #pragma unroll
        for (int i = 0; i < 4; ++i) {
            int n0 = (w * 4 + i) * 16;
            const __bf16* tp = Tt + (n0 + l15) * K2 + k0 + hi * 8;
            BFrag b;
            b.h[0] = *(const v8bf*)(tp);
            b.h[1] = *(const v8bf*)(tp + 16);
            acc[i] = __builtin_amdgcn_wmma_f32_16x16x32_bf16(
                false, a.v, false, b.v, (short)0, acc[i], false, false);
        }
    }

    #pragma unroll
    for (int i = 0; i < 4; ++i) {
        int col = (w * 4 + i) * 16 + l15;
        #pragma unroll
        for (int v = 0; v < 8; ++v)
            smlike[v + hi * 8][col] = acc[i][v] + 0.5f;
    }
    __syncthreads();

    // row sums: 16 threads per row, 32 cols each
    {
        int r = tid >> 4, c0 = tid & 15;
        float s = 0.0f;
        #pragma unroll
        for (int j = 0; j < 32; ++j) s += smlike[r][c0 + 16 * j];
        smpart[r][c0] = s;
    }
    __syncthreads();
    if (tid < 16) {
        float s = 0.0f;
        #pragma unroll
        for (int j = 0; j < 16; ++j) s += smpart[tid][j];
        sminv[tid] = 1.0f / s;
    }
    __syncthreads();

    // fully coalesced store of log(dist + 1e-5)
    float* orow = out + row0 * DOUT;
    #pragma unroll 1
    for (int idx = tid; idx < 16 * DOUT; idx += 256) {
        int r = idx >> 9, c = idx & 511;
        orow[idx] = logf(smlike[r][c] * sminv[r] + 1e-5f);
    }
}

// ---------- launch ----------
extern "C" void kernel_launch(void* const* d_in, const int* in_sizes, int n_in,
                              void* d_out, int out_size, void* d_ws, size_t ws_size,
                              hipStream_t stream) {
    const float* x = (const float*)d_in[0];   // [16,4096,1024] f32
    const float* W = (const float*)d_in[1];   // [130,1024] f32
    const float* b = (const float*)d_in[2];   // [130] f32
    float* out = (float*)d_out;               // [16,4096,512] f32

    char* ws = (char*)d_ws;
    __bf16* Wbf  = (__bf16*)(ws);                                // 144*1024*2   = 294912 B
    __bf16* Tt   = (__bf16*)(ws + 294912);                       // 512*128*2    = 131072 B
    __bf16* cmat = (__bf16*)(ws + 294912 + 131072);              // 65536*128*2  = 16 MiB

    k_prep_w   <<<(NPAD * DIN) / 256, 256, 0, stream>>>(W, Wbf);
    k_prep_trig<<<(DOUT * K2)  / 256, 256, 0, stream>>>(Tt);
    k_gemm_autocorr<<<BROWS / (8 * 16), 256, 0, stream>>>(x, Wbf, b, cmat);
    k_trig         <<<BROWS / 16,       256, 0, stream>>>(cmat, Tt, out);
}
